// RelationNetwork_88510686036078
// MI455X (gfx1250) — matrix operand order repaired and checked by reading, not compiled, not run on hardware
//
#include <hip/hip_runtime.h>

// ---------------------------------------------------------------------------
// Relation Network for MI455X (gfx1250, wave32, WMMA f16->f32)
// B=32, N=64, D=128; g: 384->256->256->256 (ReLU all), sum over pairs,
// f: 256->256 (ReLU) ->32.
// Layer-0 factorization: h0[b, j*64+i] = relu(A[b,i] + Bm[b,j] + C[b])
//   A = x_h @ W0a, Bm = x_h @ W0b, C = q @ W0c + b0   (W0 split by input rows)
// ---------------------------------------------------------------------------

typedef __attribute__((ext_vector_type(16))) _Float16 v16h;
typedef __attribute__((ext_vector_type(8)))  float    v8f;
typedef __attribute__((ext_vector_type(4)))  float    f32x4;

#define B_   32
#define N_   64
#define D_   128
#define GH_  256
#define FOUT_ 32

#define LDH (GH_ + 8)   // padded LDS leading dim (halves) for 256-wide tiles
#define LDX (D_  + 8)   // padded LDS leading dim for 128-wide x tiles

// A-fragment (16x32 f16, ISA 7.12.2): lane<16 -> row M=lane, K chunks {0-7,16-23};
// lane>=16 -> row M=lane-16 (same rows, K chunks {8-15,24-31}).
__device__ __forceinline__ v16h load_a_frag(const _Float16* t, int ld, int row,
                                            int k0, int ks) {
  const _Float16* p = t + row * ld + k0 + ks * 8;
  union { v16h v; f32x4 f[2]; } u;
  u.f[0] = *(const f32x4*)(p);        // halves 0..7   (VGPR 0-3)
  u.f[1] = *(const f32x4*)(p + 16);   // halves 8..15  (VGPR 4-7)
  return u.v;
}

// B-fragment (32x16 f16): lane = column N; lanes 0-15 hold K {0-7,16-23}, lanes 16-31
// hold K {8-15,24-31} (same as A). W stored transposed [Nout][Kin] so this
// is two contiguous 16-byte reads per lane.
__device__ __forceinline__ v16h load_b_frag(const _Float16* W, int ld, int col,
                                            int khalf) {
  const _Float16* p = W + col * ld + khalf;
  union { v16h v; f32x4 f[2]; } u;
  u.f[0] = *(const f32x4*)(p);
  u.f[1] = *(const f32x4*)(p + 16);
  return u.v;
}

// One 64x256 output tile per 256-thread block (8 waves).
// Wave w owns columns [w*32, w*32+32): B frags loaded once per WG, reused over
// all 4 M-tiles. Accumulators: 4 (M) x 2 (N) tiles of 16x16 f32.
__device__ __forceinline__ void gemm_tile(const _Float16* __restrict__ tile, int ldt,
                                          int K, const _Float16* __restrict__ WT,
                                          v8f acc[4][2]) {
  const int lane = threadIdx.x & 31;
  const int lh   = lane & 15;
  const int ks   = lane >> 4;
  const int n0   = (threadIdx.x >> 5) * 32;
  for (int kt = 0; kt < K; kt += 32) {
    v16h b0 = load_b_frag(WT, K, n0 + lh,      kt + ks * 8);
    v16h b1 = load_b_frag(WT, K, n0 + 16 + lh, kt + ks * 8);
#pragma unroll
    for (int m = 0; m < 4; ++m) {
      v16h a = load_a_frag(tile, ldt, m * 16 + lh, kt, ks);
      acc[m][0] = __builtin_amdgcn_wmma_f32_16x16x32_f16(
          false, a, false, b0, (short)0, acc[m][0], false, false);
      acc[m][1] = __builtin_amdgcn_wmma_f32_16x16x32_f16(
          false, a, false, b1, (short)0, acc[m][1], false, false);
    }
  }
}

// ---------------------------------------------------------------------------
// K0: fp32->fp16 converts, weight transposes, C = q @ W0c + b0, zero emb.
// ---------------------------------------------------------------------------
__global__ void rn_prep(const float* __restrict__ x, const float* __restrict__ q,
                        const float* __restrict__ gw0, const float* __restrict__ gb0,
                        const float* __restrict__ gw1, const float* __restrict__ gw2,
                        _Float16* __restrict__ xh,
                        _Float16* __restrict__ W0aT, _Float16* __restrict__ W0bT,
                        _Float16* __restrict__ W1T,  _Float16* __restrict__ W2T,
                        float* __restrict__ Cb, float* __restrict__ emb) {
  const int tid = blockIdx.x * blockDim.x + threadIdx.x;
  const int nt  = gridDim.x * blockDim.x;
  for (int i = tid; i < B_ * N_ * D_; i += nt) xh[i] = (_Float16)x[i];
  for (int i = tid; i < GH_ * D_; i += nt) {           // [n][k] = gw0[k][n]
    int n = i / D_, k = i % D_;
    W0aT[i] = (_Float16)gw0[k * GH_ + n];
    W0bT[i] = (_Float16)gw0[(D_ + k) * GH_ + n];
  }
  for (int i = tid; i < GH_ * GH_; i += nt) {
    int n = i / GH_, k = i % GH_;
    W1T[i] = (_Float16)gw1[k * GH_ + n];
    W2T[i] = (_Float16)gw2[k * GH_ + n];
  }
  for (int i = tid; i < B_ * GH_; i += nt) {           // C = q @ W0c + b0 (fp32)
    int b = i / GH_, c = i % GH_;
    float s = gb0[c];
    for (int k = 0; k < D_; ++k) s += q[b * D_ + k] * gw0[(2 * D_ + k) * GH_ + c];
    Cb[i]  = s;
    emb[i] = 0.0f;
  }
}

// ---------------------------------------------------------------------------
// K1: A = xh @ W0a, Bm = xh @ W0b   (M=2048, K=128, N=256), fp32 out.
// ---------------------------------------------------------------------------
__global__ void __launch_bounds__(256)
rn_layer0(const _Float16* __restrict__ xh,
          const _Float16* __restrict__ W0aT, const _Float16* __restrict__ W0bT,
          float* __restrict__ A, float* __restrict__ Bm) {
  __shared__ _Float16 xt[64 * LDX];
  const int mblk = blockIdx.x;                      // 32 blocks of 64 rows
  const _Float16* W = blockIdx.y ? W0bT : W0aT;
  float* Out        = blockIdx.y ? Bm   : A;
  for (int i = threadIdx.x; i < 64 * D_; i += 256) {
    int r = i >> 7, c = i & (D_ - 1);
    xt[r * LDX + c] = xh[(mblk * 64 + r) * D_ + c];
  }
  __syncthreads();
  v8f acc[4][2] = {};
  gemm_tile(xt, LDX, D_, W, acc);
  const int lane = threadIdx.x & 31, lh = lane & 15, ks = lane >> 4;
  const int n0   = (threadIdx.x >> 5) * 32;
#pragma unroll
  for (int m = 0; m < 4; ++m)
#pragma unroll
    for (int n = 0; n < 2; ++n) {
      int col = n0 + n * 16 + lh;
#pragma unroll
      for (int v = 0; v < 8; ++v) {
        int row = mblk * 64 + m * 16 + ks * 8 + v;   // C/D layout: VGPR v -> row
        Out[row * GH_ + col] = acc[m][n][v];
      }
    }
}

// ---------------------------------------------------------------------------
// K2: per (b,j): h0 tile (64x256) -> g layer1 -> g layer2 -> column sums ->
// atomic add into emb[b]. Single LDS tile reused h0 -> h1 (barrier-fenced).
// ---------------------------------------------------------------------------
__global__ void __launch_bounds__(256)
rn_main(const float* __restrict__ A, const float* __restrict__ Bm,
        const float* __restrict__ Cb,
        const _Float16* __restrict__ W1T, const float* __restrict__ gb1,
        const _Float16* __restrict__ W2T, const float* __restrict__ gb2,
        float* __restrict__ emb) {
  __shared__ _Float16 ht[64 * LDH];                  // 33 KB
  const int blk = blockIdx.x;                        // b*64 + j
  const int b = blk >> 6, j = blk & 63;
  const float* Ar = A  + (b * 64) * GH_;
  const float* Br = Bm + (b * 64 + j) * GH_;
  const float* Cr = Cb + b * GH_;
  for (int i = threadIdx.x; i < 64 * GH_; i += 256) {
    int r = i >> 8, c = i & 255;
    float v = Ar[r * GH_ + c] + Br[c] + Cr[c];
    ht[r * LDH + c] = (_Float16)fmaxf(v, 0.0f);      // h0 = relu(A+Bm+C)
  }
  __syncthreads();

  const int lane = threadIdx.x & 31, lh = lane & 15, ks = lane >> 4;
  const int n0   = (threadIdx.x >> 5) * 32;

  v8f acc[4][2] = {};
  gemm_tile(ht, LDH, GH_, W1T, acc);                 // layer 1
  __syncthreads();                                   // everyone done reading h0
#pragma unroll
  for (int m = 0; m < 4; ++m)
#pragma unroll
    for (int n = 0; n < 2; ++n) {
      int col = n0 + n * 16 + lh;
      float bias = gb1[col];
#pragma unroll
      for (int v = 0; v < 8; ++v)
        ht[(m * 16 + ks * 8 + v) * LDH + col] =
            (_Float16)fmaxf(acc[m][n][v] + bias, 0.0f);   // h1
    }
  __syncthreads();

  const v8f vz = {};
#pragma unroll
  for (int m = 0; m < 4; ++m) { acc[m][0] = vz; acc[m][1] = vz; }
  gemm_tile(ht, LDH, GH_, W2T, acc);                 // layer 2

#pragma unroll
  for (int n = 0; n < 2; ++n) {                      // relu + reduce 64 rows
    int col = n0 + n * 16 + lh;
    float bias = gb2[col];
    float s = 0.0f;
#pragma unroll
    for (int m = 0; m < 4; ++m)
#pragma unroll
      for (int v = 0; v < 8; ++v)
        s += fmaxf(acc[m][n][v] + bias, 0.0f);
    s += __shfl_xor(s, 16, 32);                      // join the two row halves
    if (ks == 0) atomicAdd(&emb[b * GH_ + col], s);  // global_atomic_add_f32
  }
}

// ---------------------------------------------------------------------------
// K3: f MLP, fp32 scalar (tiny): out = relu(emb@fw0+fb0) @ fw1 + fb1
// ---------------------------------------------------------------------------
__global__ void __launch_bounds__(256)
rn_fmlp(const float* __restrict__ emb,
        const float* __restrict__ fw0, const float* __restrict__ fb0,
        const float* __restrict__ fw1, const float* __restrict__ fb1,
        float* __restrict__ out) {
  __shared__ float h[B_ * GH_];
  for (int i = threadIdx.x; i < B_ * GH_; i += 256) {
    int b = i >> 8, c = i & 255;
    float s = fb0[c];
    for (int k = 0; k < GH_; ++k) s += emb[b * GH_ + k] * fw0[k * GH_ + c];
    h[i] = fmaxf(s, 0.0f);
  }
  __syncthreads();
  for (int i = threadIdx.x; i < B_ * FOUT_; i += 256) {
    int b = i >> 5, o = i & 31;
    float s = fb1[o];
    for (int c = 0; c < GH_; ++c) s += h[b * GH_ + c] * fw1[c * FOUT_ + o];
    out[i] = s;
  }
}

// ---------------------------------------------------------------------------
extern "C" void kernel_launch(void* const* d_in, const int* in_sizes, int n_in,
                              void* d_out, int out_size, void* d_ws, size_t ws_size,
                              hipStream_t stream) {
  (void)in_sizes; (void)n_in; (void)out_size; (void)ws_size;
  const float* x   = (const float*)d_in[0];
  const float* q   = (const float*)d_in[1];
  const float* gw0 = (const float*)d_in[2];
  const float* gb0 = (const float*)d_in[3];
  const float* gw1 = (const float*)d_in[4];
  const float* gb1 = (const float*)d_in[5];
  const float* gw2 = (const float*)d_in[6];
  const float* gb2 = (const float*)d_in[7];
  const float* fw0 = (const float*)d_in[8];
  const float* fb0 = (const float*)d_in[9];
  const float* fw1 = (const float*)d_in[10];
  const float* fb1 = (const float*)d_in[11];
  float* out = (float*)d_out;

  char* ws = (char*)d_ws;
  size_t off = 0;
  auto alloc = [&](size_t bytes) -> char* {
    char* p = ws + off;
    off += (bytes + 255) & ~(size_t)255;
    return p;
  };
  _Float16* xh   = (_Float16*)alloc((size_t)B_ * N_ * D_ * 2);   // 512 KB
  _Float16* W0aT = (_Float16*)alloc((size_t)GH_ * D_ * 2);       // 64 KB
  _Float16* W0bT = (_Float16*)alloc((size_t)GH_ * D_ * 2);
  _Float16* W1T  = (_Float16*)alloc((size_t)GH_ * GH_ * 2);      // 128 KB
  _Float16* W2T  = (_Float16*)alloc((size_t)GH_ * GH_ * 2);
  float*    Abuf = (float*)alloc((size_t)B_ * N_ * GH_ * 4);     // 2 MB
  float*    Bbuf = (float*)alloc((size_t)B_ * N_ * GH_ * 4);     // 2 MB
  float*    Cb   = (float*)alloc((size_t)B_ * GH_ * 4);
  float*    emb  = (float*)alloc((size_t)B_ * GH_ * 4);

  rn_prep<<<256, 256, 0, stream>>>(x, q, gw0, gb0, gw1, gw2,
                                   xh, W0aT, W0bT, W1T, W2T, Cb, emb);
  rn_layer0<<<dim3(32, 2), 256, 0, stream>>>(xh, W0aT, W0bT, Abuf, Bbuf);
  rn_main<<<B_ * N_, 256, 0, stream>>>(Abuf, Bbuf, Cb, W1T, gb1, W2T, gb2, emb);
  rn_fmlp<<<1, 256, 0, stream>>>(emb, fw0, fb0, fw1, fb1, out);
}
